// KPConv_42099269435605
// MI455X (gfx1250) — matrix operand order repaired and checked by reading, not compiled, not run on hardware
//
#include <hip/hip_runtime.h>

typedef __attribute__((ext_vector_type(16))) _Float16 v16h;
typedef __attribute__((ext_vector_type(8)))  _Float16 v8h;
typedef __attribute__((ext_vector_type(8)))  float    v8f;

#define B_     4
#define N_     8192
#define CIN_   64
#define COUT_  128
#define P_     15
#define K_     32
#define EPS_   1e-5f
#define KTOT   960      // P_*CIN_ (GEMM reduction dim)
#define NCHUNK 30       // KTOT/32
#define FROW   1032     // feat row stride (halves): [960,1032) is a dead pad
                        // region absorbing the p=15 WMMA pad row (no guard)
#define CSTR   40       // corrA row stride (halves): 80B rows, conflict-free
#define GSTR   40       // gpT  row stride (halves)

// ---------------------------------------------------------------------------
// Kernel 0: swizzle weights (P,CIN,COUT) f32 -> f16 B-fragment-ready layout.
// W3[kc][ot][lane][e]: lane 0-15: col = ot*16+lane,    K = kc*32+e
//                      lane16-31: col = ot*16+lane-16, K = kc*32+16+e
// => each GEMM wave loads one contiguous v16h (32B) per (kc, ot).
// ---------------------------------------------------------------------------
__global__ void prep_w3(const float* __restrict__ w, _Float16* __restrict__ W3) {
    int id = blockIdx.x * 256 + threadIdx.x;
    if (id >= NCHUNK * 8 * 32 * 16) return;
    int e    = id & 15;
    int lane = (id >> 4) & 31;
    int ot   = (id >> 9) & 7;
    int kc   = id >> 12;
    int col  = ot * 16 + (lane & 15);
    int krow = kc * 32 + ((lane >> 4) << 4) + e;   // K index in [0,960)
    W3[id] = (_Float16)w[krow * COUT_ + col];
}

// ---------------------------------------------------------------------------
// LDS max-heap (32 entries, u64 = dist_bits<<32 | idx). Root is the running
// threshold, cached in a register. Dist >= 0 so float-bit order == value order.
// Heap slot s for thread tid lives at hp[(s<<8)+tid] -> wave b64 accesses to a
// fixed slot span exactly 64 banks (conflict-free).
// ---------------------------------------------------------------------------
__device__ __forceinline__ void heap_insert(unsigned long long* hp, int tid,
                                            unsigned long long e,
                                            unsigned long long& root) {
    unsigned long long v = e;
    int s = 0;
    bool done = false;
#pragma unroll
    for (int lvl = 0; lvl < 5; ++lvl) {          // depth of 32-heap is 5
        if (!done) {
            const int c1 = 2 * s + 1, c2 = 2 * s + 2;
            if (c1 < 32) {
                const unsigned long long l1 = hp[(c1 << 8) + tid];
                const unsigned long long l2 = (c2 < 32) ? hp[(c2 << 8) + tid] : 0ULL;
                const bool pick1 = l1 > l2;
                const unsigned long long m = pick1 ? l1 : l2;
                const int cm = pick1 ? c1 : c2;
                if (m > v) { hp[(s << 8) + tid] = m; s = cm; }
                else       { hp[(s << 8) + tid] = v; done = true; }
            } else       { hp[(s << 8) + tid] = v; done = true; }
        }
    }
    if (!done) hp[(s << 8) + tid] = v;
    root = hp[tid];                               // new max (slot 0)
}

// ---------------------------------------------------------------------------
// Kernel 1: self-KNN, K=32, split-candidate. 256-thread block = 128 queries x
// 2 segments of 4096 candidates (2x wave residency, half-length dependency
// chains). Unsorted top-K is fine: downstream aggregation sums over k.
// ---------------------------------------------------------------------------
__global__ void __launch_bounds__(256) knn_kernel(const float* __restrict__ xyz,
                                                  int* __restrict__ idxOut) {
    extern __shared__ char kmem[];
    unsigned long long* hp = (unsigned long long*)kmem;   // 256*32*8 = 64 KB
    float* sx = (float*)(kmem + 65536);                   // 256 floats
    float* sy = sx + 256;
    float* sz = sy + 256;

    const int tid = threadIdx.x;
    const int q   = tid & 127;                // query slot within block
    const int seg = tid >> 7;                 // candidate segment 0/1
    const int b   = blockIdx.x >> 6;          // 64 blocks per batch
    const int n   = ((blockIdx.x & 63) << 7) + q;
    const float* xb = xyz + (size_t)b * N_ * 3;
    const float qx = xb[n * 3 + 0];
    const float qy = xb[n * 3 + 1];
    const float qz = xb[n * 3 + 2];

    const unsigned long long SENT = 0x7F7FFFFF00000000ULL;   // FLT_MAX | 0
#pragma unroll
    for (int s = 0; s < 32; ++s) hp[(s << 8) + tid] = SENT;
    unsigned long long root = SENT;

    const int cbase = seg * (N_ / 2);
    const int tb = seg << 7;
    for (int t = 0; t < N_ / 2; t += 128) {
        const int ci = cbase + t + q;         // coalesced per segment half
        sx[tid] = xb[ci * 3 + 0];
        sy[tid] = xb[ci * 3 + 1];
        sz[tid] = xb[ci * 3 + 2];
        __syncthreads();
        for (int u = 0; u < 128; ++u) {
            const float dx = sx[tb + u] - qx;
            const float dy = sy[tb + u] - qy;
            const float dz = sz[tb + u] - qz;
            const float d = dx * dx + dy * dy + dz * dz;
            const unsigned long long e =
                ((unsigned long long)__float_as_uint(d) << 32)
                | (unsigned)(cbase + t + u);
            if (e < root) heap_insert(hp, tid, e, root);
        }
        __syncthreads();
    }

    // merge: segment 0 absorbs segment 1's 32 entries through its own heap
    if (seg == 0) {
        for (int s2 = 0; s2 < 32; ++s2) {
            const unsigned long long e = hp[(s2 << 8) + tid + 128];
            if (e < root) heap_insert(hp, tid, e, root);
        }
        const int base = ((b * N_) + n) * K_;
        for (int s2 = 0; s2 < 32; ++s2)
            idxOut[base + s2] = (int)(hp[(s2 << 8) + tid] & 0xFFFFFFFFu);
    }
}

// ---------------------------------------------------------------------------
// Kernel 2: fused correlation + feat-WMMA + GEMM-WMMA.
// One workgroup (8 waves) per 16-point tile.
// Stage 1 (per wave, 2 points): corr -> LDS [p][k] f16 (A-frag layout),
//   gathered neighbor feats -> LDS transposed gpT[c][k] f16 (B-frag layout),
//   feat(16x64) = corrT(16x32) @ gp(32x64) via 4x v_wmma, D written straight
//   into the stage-2 A-tile (featA, f16; p=15 pad row lands in dead region).
// Stage 2: out tile (16x128) = featA(16x960) @ W3(960x128), 30 WMMAs/wave.
// ---------------------------------------------------------------------------
__global__ void __launch_bounds__(256) kpconv_kernel(
        const float* __restrict__ xyz, const float* __restrict__ points,
        const float* __restrict__ kp, const int* __restrict__ idx,
        const _Float16* __restrict__ W3, float* __restrict__ out) {
    extern __shared__ char smem[];
    int*      idxW  = (int*)smem;                         //  8*32*4        = 1 KB
    _Float16* corrA = (_Float16*)(smem + 1024);           //  8*16*CSTR*2   = 10 KB
    _Float16* gpT   = (_Float16*)(smem + 11264);          //  8*64*GSTR*2   = 40 KB
    _Float16* featA = (_Float16*)(smem + 52224);          // 16*FROW*2      = 32.25 KB

    const int b   = blockIdx.x >> 9;         // N_/16 = 512 tiles per batch
    const int n0  = (blockIdx.x & 511) << 4;
    const int tid = threadIdx.x;
    const int w   = tid >> 5;                // wave id 0..7
    const int j   = tid & 31;                // lane id

    const int half8  = (j >> 4) << 3;        // 0 / 8
    const int half16 = (j >> 4) << 4;        // 0 / 16
    const int l15    = j & 15;

    // warm the L2->WGP path for this wave's W3 stream (global_prefetch_b8)
    __builtin_prefetch(W3 + ((size_t)w * 32 + j) * 16, 0, 1);

    // ------------------ Stage 1: build A-tile (feat, f16) ------------------
    for (int sub = 0; sub < 2; ++sub) {
        const int m = w * 2 + sub;           // tile row (point) this wave builds
        const int n = n0 + m;
        const int nbr = idx[(((size_t)b * N_) + n) * K_ + j];
        idxW[w * 32 + j] = nbr;

        const float qx = xyz[((size_t)b * N_ + n) * 3 + 0];
        const float qy = xyz[((size_t)b * N_ + n) * 3 + 1];
        const float qz = xyz[((size_t)b * N_ + n) * 3 + 2];
        const float nx = xyz[((size_t)b * N_ + nbr) * 3 + 0];
        const float ny = xyz[((size_t)b * N_ + nbr) * 3 + 1];
        const float nz = xyz[((size_t)b * N_ + nbr) * 3 + 2];
        const float rx = (nx - qx) * 2.0f;   // / RADIUS (0.5)
        const float ry = (ny - qy) * 2.0f;
        const float rz = (nz - qz) * 2.0f;

        // corr[k=j][p] -> corrA[p][k] (A-matrix element A[p][k])
#pragma unroll
        for (int p = 0; p < P_; ++p) {
            const float dx = rx - kp[p * 3 + 0];
            const float dy = ry - kp[p * 3 + 1];
            const float dz = rz - kp[p * 3 + 2];
            const float c  = 1.0f - sqrtf(dx * dx + dy * dy + dz * dz);
            corrA[(w * 16 + p) * CSTR + j] = (_Float16)(c > 0.0f ? c : 0.0f);
        }
        corrA[(w * 16 + 15) * CSTR + j] = (_Float16)0.0f;   // pad row p=15
        __syncthreads();

        // gather neighbor features (coalesced global rows), store transposed
        for (int r = 0; r < K_; ++r) {
            const int ii = idxW[w * 32 + r];
            const float* prow = points + ((size_t)b * N_ + ii) * CIN_;
            gpT[(w * 64 + j) * GSTR + r]        = (_Float16)prow[j];
            gpT[(w * 64 + j + 32) * GSTR + r]   = (_Float16)prow[j + 32];
        }
        __syncthreads();

        // A fragment (corr): lane row p = j%16, K = half8+{0..7}, half8+16+{0..7}
        const _Float16* ap = corrA + (w * 16 + l15) * CSTR + half8;
        const v8h alo = *(const v8h*)ap;
        const v8h ahi = *(const v8h*)(ap + 16);
        v16h a;
#pragma unroll
        for (int e = 0; e < 8; ++e) { a[e] = alo[e]; a[e + 8] = ahi[e]; }

        // hoist all 4 B fragments (one dscnt wait covers them)
        v16h bfr[4];
#pragma unroll
        for (int ct = 0; ct < 4; ++ct) {
            const _Float16* bp = gpT + (w * 64 + ct * 16 + l15) * GSTR + half8;
            const v8h blo = *(const v8h*)bp;
            const v8h bhi = *(const v8h*)(bp + 16);
#pragma unroll
            for (int e = 0; e < 8; ++e) { bfr[ct][e] = blo[e]; bfr[ct][e + 8] = bhi[e]; }
        }

        // feat = corrT @ gp : 4 WMMAs, D stored straight into stage-2 A-tile
#pragma unroll
        for (int ct = 0; ct < 4; ++ct) {
            v8f cz = {};
            const v8f d = __builtin_amdgcn_wmma_f32_16x16x32_f16(
                              false, a, false, bfr[ct], (short)0, cz, false, false);
#pragma unroll
            for (int v = 0; v < 8; ++v)
                featA[m * FROW + (v + half8) * 64 + ct * 16 + l15] = (_Float16)d[v];
        }
        __syncthreads();
    }

    // ------------- Stage 2: (16x960)@(960x128) -> 16x128 output -------------
    const int ot = w;                        // 16-col tile per wave
    v8f acc = {};
#pragma unroll
    for (int kc = 0; kc < NCHUNK; ++kc) {
        const _Float16* ap = featA + l15 * FROW + kc * 32 + half16;
        const v8h lo = *(const v8h*)ap;          // K = half16 .. +7
        const v8h hi = *(const v8h*)(ap + 8);    // K = half16+8 .. +15
        v16h a;
#pragma unroll
        for (int e = 0; e < 8; ++e) { a[e] = lo[e]; a[e + 8] = hi[e]; }
        const v16h bf = *(const v16h*)(W3 + (((size_t)kc * 8 + ot) * 32 + j) * 16);
        acc = __builtin_amdgcn_wmma_f32_16x16x32_f16(
                  false, a, false, bf, (short)0, acc, false, false);
    }
    const int col = ot * 16 + l15;
#pragma unroll
    for (int v = 0; v < 8; ++v) {
        out[(((size_t)b * N_) + n0 + half8 + v) * COUT_ + col] = acc[v];
    }
}

// ---------------------------------------------------------------------------
// Kernel 3: BN batch stats -> fused scale/shift per channel.
// ---------------------------------------------------------------------------
__global__ void __launch_bounds__(256) bn_stats(const float* __restrict__ out,
                                                const float* __restrict__ bnw,
                                                const float* __restrict__ bnb,
                                                float* __restrict__ stats) {
    __shared__ float ssum[256], ssq[256];
    const int o = blockIdx.x;
    float s = 0.f, q = 0.f;
    for (int r = threadIdx.x; r < B_ * N_; r += 256) {
        const float v = out[(size_t)r * COUT_ + o];
        s += v; q += v * v;
    }
    ssum[threadIdx.x] = s; ssq[threadIdx.x] = q;
    __syncthreads();
    for (int st = 128; st > 0; st >>= 1) {
        if (threadIdx.x < st) {
            ssum[threadIdx.x] += ssum[threadIdx.x + st];
            ssq[threadIdx.x]  += ssq[threadIdx.x + st];
        }
        __syncthreads();
    }
    if (threadIdx.x == 0) {
        const float inv  = 1.0f / (float)(B_ * N_);
        const float mean = ssum[0] * inv;
        const float var  = ssq[0] * inv - mean * mean;
        const float sc   = bnw[o] * rsqrtf(var + EPS_);
        stats[o]         = sc;
        stats[COUT_ + o] = bnb[o] - mean * sc;
    }
}

// ---------------------------------------------------------------------------
// Kernel 4: apply BN in place.
// ---------------------------------------------------------------------------
__global__ void __launch_bounds__(256) bn_apply(float* __restrict__ out,
                                                const float* __restrict__ stats) {
    const int id = blockIdx.x * 256 + threadIdx.x;
    const int o = id & (COUT_ - 1);
    out[id] = out[id] * stats[o] + stats[COUT_ + o];
}

// ---------------------------------------------------------------------------
extern "C" void kernel_launch(void* const* d_in, const int* in_sizes, int n_in,
                              void* d_out, int out_size, void* d_ws, size_t ws_size,
                              hipStream_t stream) {
    const float* xyz     = (const float*)d_in[0];   // (B,N,3)
    const float* points  = (const float*)d_in[1];   // (B,N,CIN)
    const float* kp      = (const float*)d_in[2];   // (P,3)
    const float* weights = (const float*)d_in[3];   // (P,CIN,COUT)
    const float* bnw     = (const float*)d_in[4];   // (COUT,)
    const float* bnb     = (const float*)d_in[5];   // (COUT,)
    float* out = (float*)d_out;                     // (B,N,COUT)

    char* ws = (char*)d_ws;
    int*      idx   = (int*)ws;                                         // 4 MiB
    _Float16* W3    = (_Float16*)(ws + (size_t)B_ * N_ * K_ * 4);       // 240 KiB
    float*    stats = (float*)(ws + (size_t)B_ * N_ * K_ * 4
                                  + (size_t)NCHUNK * 4096 * 2);         // 1 KiB

    prep_w3<<<(NCHUNK * 4096 + 255) / 256, 256, 0, stream>>>(weights, W3);

    const size_t kmem = 65536 + 3 * 256 * 4;                            // 67 KB
    knn_kernel<<<B_ * (N_ / 128), 256, kmem, stream>>>(xyz, idx);

    const size_t smem = 52224 + (size_t)16 * FROW * 2;                  // ~83 KB
    kpconv_kernel<<<B_ * (N_ / 16), 256, smem, stream>>>(xyz, points, kp, idx, W3, out);

    bn_stats<<<COUT_, 256, 0, stream>>>(out, bnw, bnb, stats);
    bn_apply<<<(B_ * N_ * COUT_) / 256, 256, 0, stream>>>(out, stats);
}
